// MoDL_83013127897296
// MI455X (gfx1250) — compile-verified, matrix operand order checked by reading, not run on hardware
//
#include <hip/hip_runtime.h>

// ---------------------------------------------------------------------------
// MoDL: 5-layer CNN denoiser (bf16 WMMA implicit GEMM for the 64->64 convs)
// + 11-iteration CG with per-coil 2D FFTs (fp32, LDS radix-2, L2 resident).
// Workspace requirement: ~136 MB (1MB header + 2 x 64MB activation regions).
// ---------------------------------------------------------------------------

#define BB 8
#define CC 12
#define HH 256
#define WW 256
#define NPIX (BB * HH * WW)   // 524288
#define CG_TOL 1e-10f

typedef __attribute__((ext_vector_type(16))) __bf16         v16bf;
typedef __attribute__((ext_vector_type(8)))  float          v8f;
typedef __attribute__((ext_vector_type(8)))  unsigned short ush8;
typedef __attribute__((ext_vector_type(16))) unsigned short ush16;

__device__ __forceinline__ float bf2f(unsigned short h) {
    return __uint_as_float(((unsigned)h) << 16);
}
__device__ __forceinline__ unsigned short f2bf(float f) {
    unsigned u = __float_as_uint(f);
    u += 0x7FFFu + ((u >> 16) & 1u);          // round-to-nearest-even
    return (unsigned short)(u >> 16);
}
__device__ __forceinline__ int rev8(int v) {
    return (int)(__brev((unsigned)v) >> 24);
}

// ---------------------------------------------------------------------------
// Weight repack: fp32 OIHW [64][64][3][3] -> bf16 WMMA B-fragments
// layout [kstep(18)][ntile(4)][lane(32)][half(16)], K = (ky*3+kx)*64 + c.
// ---------------------------------------------------------------------------
__global__ void repack_w(const float* __restrict__ w2, const float* __restrict__ w3,
                         const float* __restrict__ w4, unsigned short* __restrict__ wrep)
{
    int idx = blockIdx.x * 256 + threadIdx.x;
    if (idx >= 3 * 36864) return;
    int l = idx / 36864;
    int w = idx - l * 36864;
    int hh   = w & 15;
    int lane = (w >> 4) & 31;
    int n    = (w >> 9) & 3;
    int s    = w >> 11;
    int base = (hh < 8) ? hh : hh + 8;
    int K = s * 32 + base + ((lane >= 16) ? 8 : 0);
    int N = n * 16 + (lane & 15);
    int t = K >> 6, c = K & 63;
    int ky = t / 3, kx = t - ky * 3;
    const float* wsrc = (l == 0) ? w2 : (l == 1) ? w3 : w4;
    wrep[idx] = f2bf(wsrc[((N * 64 + c) * 3 + ky) * 3 + kx]);
}

// ---------------------------------------------------------------------------
// conv1: 2 -> 64, ReLU, fp32 NCHW in -> bf16 NHWC out (K=18, direct)
// ---------------------------------------------------------------------------
__global__ __launch_bounds__(256)
void conv1_in(const float* __restrict__ atb, const float* __restrict__ w1,
              const float* __restrict__ b1, unsigned short* __restrict__ actout)
{
    __shared__ float sw[1152];
    __shared__ float sb[64];
    for (int i = threadIdx.x; i < 1152; i += 256) sw[i] = w1[i];
    if (threadIdx.x < 64) sb[threadIdx.x] = b1[threadIdx.x];
    __syncthreads();

    const int idx = blockIdx.x * 256 + threadIdx.x;
    const int b = idx >> 16, yx = idx & 65535, y = yx >> 8, x = yx & 255;

    float in[2][3][3];
    #pragma unroll
    for (int cc = 0; cc < 2; ++cc)
        #pragma unroll
        for (int ky = 0; ky < 3; ++ky)
            #pragma unroll
            for (int kx = 0; kx < 3; ++kx) {
                int yy = y + ky - 1, xx = x + kx - 1;
                in[cc][ky][kx] = ((unsigned)yy < HH && (unsigned)xx < WW)
                    ? atb[(((size_t)b * 2 + cc) << 16) + (size_t)yy * WW + xx] : 0.f;
            }
    unsigned short* op = &actout[(size_t)idx * 64];
    for (int o = 0; o < 64; ++o) {
        float s = sb[o];
        #pragma unroll
        for (int cc = 0; cc < 2; ++cc)
            #pragma unroll
            for (int ky = 0; ky < 3; ++ky)
                #pragma unroll
                for (int kx = 0; kx < 3; ++kx)
                    s += in[cc][ky][kx] * sw[(o * 2 + cc) * 9 + ky * 3 + kx];
        op[o] = f2bf(fmaxf(s, 0.f));
    }
}

// ---------------------------------------------------------------------------
// conv64: 64 -> 64 (+ReLU), bf16 NHWC, implicit GEMM via v_wmma_f32_16x16x32_bf16
// Block: 256 thr (8 waves). Each wave: 16 pixels x 64 channels -> 4 acc tiles,
// 18 K-steps -> 72 WMMA per wave. A tile (3 rows x 130 x 64 ch) staged in LDS.
// ---------------------------------------------------------------------------
__global__ __launch_bounds__(256)
void conv64_wmma(const unsigned short* __restrict__ actin,
                 const unsigned short* __restrict__ bfrag,
                 const float* __restrict__ bias,
                 unsigned short* __restrict__ actout,
                 int relu)
{
    extern __shared__ __align__(16) unsigned short lds[];  // 3*130*64 bf16
    const int tid = threadIdx.x;
    const int b = blockIdx.z, y = blockIdx.y, x0 = blockIdx.x * 128;

    // Stage halo tile: [row 0..2][xi 0..129][c 0..63], OOB -> 0
    for (int p = tid; p < 3 * 130; p += 256) {
        const int row = p / 130, xi = p - row * 130;
        const int yy = y + row - 1, xx = x0 + xi - 1;
        uint4* dst = (uint4*)&lds[p * 64];
        if ((unsigned)yy < HH && (unsigned)xx < WW) {
            const uint4* src = (const uint4*)&actin[(((size_t)b * HH + yy) * WW + xx) * 64];
            #pragma unroll
            for (int q = 0; q < 8; ++q) dst[q] = src[q];
        } else {
            const uint4 z = make_uint4(0, 0, 0, 0);
            #pragma unroll
            for (int q = 0; q < 8; ++q) dst[q] = z;
        }
    }
    __syncthreads();

    const int lane = tid & 31;
    const int wv   = tid >> 5;      // wave id: pixel sub-tile
    const int mlo  = lane & 15;     // A/C: pixel row within tile / out channel
    const int hi   = lane >> 4;     // K-half select

    v8f acc[4] = {};

    for (int s = 0; s < 18; ++s) {
        union { ush16 v; ush8 h[2]; } au;
        #pragma unroll
        for (int g = 0; g < 2; ++g) {
            const int Kb = s * 32 + g * 16 + hi * 8;   // 8 contiguous K
            const int t  = Kb >> 6;
            const int c0 = Kb & 63;
            const int ky = t / 3, kx = t - ky * 3;
            const int xi = wv * 16 + mlo + kx;         // (+1 halo, -1 kernel) cancel
            au.h[g] = *(const ush8*)&lds[(ky * 130 + xi) * 64 + c0];
        }
        const v16bf a = __builtin_bit_cast(v16bf, au.v);
        #pragma unroll
        for (int n = 0; n < 4; ++n) {
            const ush16 bu = *(const ush16*)&bfrag[((size_t)((s * 4 + n) * 32 + lane)) * 16];
            const v16bf bm = __builtin_bit_cast(v16bf, bu);
            acc[n] = __builtin_amdgcn_wmma_f32_16x16x32_bf16(
                false, a, false, bm, (short)0, acc[n], false, false);
        }
    }

    // Epilogue: bias + ReLU + bf16 store (C/D layout: VGPR r -> M=r + hi*8, N=mlo)
    const int xb = x0 + wv * 16;
    #pragma unroll
    for (int n = 0; n < 4; ++n) {
        const int ch = n * 16 + mlo;
        const float bv = bias[ch];
        #pragma unroll
        for (int r = 0; r < 8; ++r) {
            float v = acc[n][r] + bv;
            if (relu) v = fmaxf(v, 0.f);
            actout[(((size_t)b * HH + y) * WW + (xb + r + hi * 8)) * 64 + ch] = f2bf(v);
        }
    }
}

// ---------------------------------------------------------------------------
// conv5 (64->2, no ReLU) + residual + rhs = (1+lam)*atb + lam*h  -> complex RHS
// ---------------------------------------------------------------------------
__global__ __launch_bounds__(256)
void conv5_rhs(const unsigned short* __restrict__ act, const float* __restrict__ atb,
               const float* __restrict__ w5, const float* __restrict__ b5,
               const float* __restrict__ lam, float2* __restrict__ rhs)
{
    __shared__ float sw[1152];
    __shared__ float sb[2];
    for (int i = threadIdx.x; i < 1152; i += 256) sw[i] = w5[i];
    if (threadIdx.x < 2) sb[threadIdx.x] = b5[threadIdx.x];
    __syncthreads();

    const int idx = blockIdx.x * 256 + threadIdx.x;
    const int b = idx >> 16, yx = idx & 65535, y = yx >> 8, x = yx & 255;

    float h0 = sb[0], h1 = sb[1];
    #pragma unroll
    for (int ky = 0; ky < 3; ++ky) {
        const int yy = y + ky - 1;
        if ((unsigned)yy >= HH) continue;
        #pragma unroll
        for (int kx = 0; kx < 3; ++kx) {
            const int xx = x + kx - 1;
            if ((unsigned)xx >= WW) continue;
            const unsigned short* ap = &act[(((size_t)b * HH + yy) * WW + xx) * 64];
            const int ko = ky * 3 + kx;
            for (int c = 0; c < 64; ++c) {
                const float v = bf2f(ap[c]);
                h0 += v * sw[c * 9 + ko];
                h1 += v * sw[(64 + c) * 9 + ko];
            }
        }
    }
    const float l = lam[0];
    const float a0 = atb[(((size_t)b * 2 + 0) << 16) + yx];
    const float a1 = atb[(((size_t)b * 2 + 1) << 16) + yx];
    rhs[idx] = make_float2((1.f + l) * a0 + l * h0, (1.f + l) * a1 + l * h1);
}

// ---------------------------------------------------------------------------
// 256-pt radix-2 DIT FFT stages in LDS. One wave per row, 4 butterflies/lane.
// Data must be pre-stored bit-reversed. sign=-1 fwd, +1 inv (unnormalized).
// ---------------------------------------------------------------------------
__device__ __forceinline__ void fft_stages(float2* row, int lane, float sign)
{
    for (int s = 1; s <= 8; ++s) {
        const int mh = 1 << (s - 1);
        __syncthreads();
        #pragma unroll
        for (int jj = 0; jj < 4; ++jj) {
            const int j = lane + jj * 32;
            const int t = j & (mh - 1);
            const int base = ((j >> (s - 1)) << s) + t;
            float sv, cv;
            __sincosf(sign * 6.28318530717958647692f * (float)t / (float)(mh << 1), &sv, &cv);
            const float2 u = row[base];
            const float2 v = row[base + mh];
            const float2 vw = make_float2(v.x * cv - v.y * sv, v.x * sv + v.y * cv);
            row[base]      = make_float2(u.x + vw.x, u.y + vw.y);
            row[base + mh] = make_float2(u.x - vw.x, u.y - vw.y);
        }
    }
    __syncthreads();
}

// coil = csm * p, then forward row-FFT
__global__ __launch_bounds__(256)
void coil_rowfft(const float2* __restrict__ P, const float* __restrict__ csr,
                 const float* __restrict__ csi, float2* __restrict__ coil)
{
    __shared__ float2 sm[8][256];
    const int lane = threadIdx.x & 31, rl = threadIdx.x >> 5;
    const int y = blockIdx.x * 8 + rl, c = blockIdx.y, b = blockIdx.z;
    float2* row = sm[rl];
    const size_t pbase = ((size_t)b << 16) + (size_t)y * WW;
    const size_t cbase = (((size_t)b * CC + c) << 16) + (size_t)y * WW;
    #pragma unroll
    for (int i = 0; i < 8; ++i) {
        const int x = lane + i * 32;
        const float2 pv = P[pbase + x];
        const float cr = csr[cbase + x], ci = csi[cbase + x];
        row[rev8(x)] = make_float2(cr * pv.x - ci * pv.y, cr * pv.y + ci * pv.x);
    }
    fft_stages(row, lane, -1.0f);
    #pragma unroll
    for (int i = 0; i < 8; ++i) { const int x = lane + i * 32; coil[cbase + x] = row[x]; }
}

// forward column FFT, mask multiply, inverse column FFT (in one pass per column)
__global__ __launch_bounds__(256)
void colfft_mask(float2* __restrict__ coil, const float* __restrict__ mask)
{
    __shared__ float2 sm[8][256];
    const int lane = threadIdx.x & 31, rl = threadIdx.x >> 5;
    const int x = blockIdx.x * 8 + rl, c = blockIdx.y, b = blockIdx.z;
    float2* row = sm[rl];
    const size_t cbase = (((size_t)b * CC + c) << 16);
    #pragma unroll
    for (int i = 0; i < 8; ++i) {
        const int y = lane + i * 32;
        row[rev8(y)] = coil[cbase + (size_t)y * WW + x];
    }
    fft_stages(row, lane, -1.0f);
    float2 tmp[8];
    #pragma unroll
    for (int i = 0; i < 8; ++i) {
        const int y = lane + i * 32;
        const float m = mask[((size_t)b << 16) + (size_t)y * WW + x];
        const float2 v = row[y];
        tmp[i] = make_float2(v.x * m, v.y * m);
    }
    __syncthreads();
    #pragma unroll
    for (int i = 0; i < 8; ++i) { const int y = lane + i * 32; row[rev8(y)] = tmp[i]; }
    fft_stages(row, lane, 1.0f);
    #pragma unroll
    for (int i = 0; i < 8; ++i) {
        const int y = lane + i * 32;
        coil[cbase + (size_t)y * WW + x] = row[y];
    }
}

// inverse row FFT per coil, accumulate conj(csm)*ci over coils, + lam*p -> Ap
__global__ __launch_bounds__(256)
void rowifft_accum(const float2* __restrict__ coil, const float* __restrict__ csr,
                   const float* __restrict__ csi, const float2* __restrict__ P,
                   const float* __restrict__ lam, float2* __restrict__ Ap)
{
    __shared__ float2 sm[8][256];
    const int lane = threadIdx.x & 31, rl = threadIdx.x >> 5;
    const int y = blockIdx.x * 8 + rl, b = blockIdx.y;
    float2* row = sm[rl];
    float2 acc[8];
    #pragma unroll
    for (int i = 0; i < 8; ++i) acc[i] = make_float2(0.f, 0.f);

    for (int c = 0; c < CC; ++c) {
        const size_t cbase = (((size_t)b * CC + c) << 16) + (size_t)y * WW;
        __syncthreads();
        #pragma unroll
        for (int i = 0; i < 8; ++i) {
            const int x = lane + i * 32;
            row[rev8(x)] = coil[cbase + x];
        }
        fft_stages(row, lane, 1.0f);
        #pragma unroll
        for (int i = 0; i < 8; ++i) {
            const int x = lane + i * 32;
            const float cr = csr[cbase + x], ci = csi[cbase + x];
            const float2 v = row[x];
            acc[i].x += cr * v.x + ci * v.y;   // conj(csm) * v
            acc[i].y += cr * v.y - ci * v.x;
        }
    }
    const float l = lam[0], inv = 1.0f / 65536.0f;   // full 2D ifft normalization
    const size_t pbase = ((size_t)b << 16) + (size_t)y * WW;
    #pragma unroll
    for (int i = 0; i < 8; ++i) {
        const int x = lane + i * 32;
        const float2 pv = P[pbase + x];
        Ap[pbase + x] = make_float2(acc[i].x * inv + l * pv.x, acc[i].y * inv + l * pv.y);
    }
}

// ---------------------------------------------------------------------------
// CG bookkeeping kernels (scalars in workspace; graph-capture safe)
// ---------------------------------------------------------------------------
__global__ void cg_init(const float2* __restrict__ rhs, float2* X, float2* R,
                        float2* Pv, int n)
{
    const int i = blockIdx.x * 256 + threadIdx.x;
    if (i >= n) return;
    X[i] = make_float2(0.f, 0.f);
    R[i] = rhs[i];
    Pv[i] = rhs[i];
}

__global__ void zero_scal(float* s, int k) { if (threadIdx.x == 0) s[k] = 0.f; }

__global__ void reduce_norm(const float2* __restrict__ v, int n, float* out)
{
    __shared__ float sb[256];
    float s = 0.f;
    for (int i = blockIdx.x * blockDim.x + threadIdx.x; i < n; i += gridDim.x * blockDim.x) {
        const float2 a = v[i];
        s += a.x * a.x + a.y * a.y;
    }
    sb[threadIdx.x] = s; __syncthreads();
    for (int o = 128; o > 0; o >>= 1) {
        if (threadIdx.x < o) sb[threadIdx.x] += sb[threadIdx.x + o];
        __syncthreads();
    }
    if (threadIdx.x == 0) atomicAdd(out, sb[0]);
}

__global__ void reduce_dot(const float2* __restrict__ a, const float2* __restrict__ b,
                           int n, float* out)
{
    __shared__ float sb[256];
    float s = 0.f;
    for (int i = blockIdx.x * blockDim.x + threadIdx.x; i < n; i += gridDim.x * blockDim.x) {
        const float2 u = a[i], v = b[i];
        s += u.x * v.x + u.y * v.y;          // Re(conj(p) * Ap)
    }
    sb[threadIdx.x] = s; __syncthreads();
    for (int o = 128; o > 0; o >>= 1) {
        if (threadIdx.x < o) sb[threadIdx.x] += sb[threadIdx.x + o];
        __syncthreads();
    }
    if (threadIdx.x == 0) atomicAdd(out, sb[0]);
}

__global__ void cg_update1(float2* X, float2* R, const float2* __restrict__ Pv,
                           const float2* __restrict__ Ap, const float* __restrict__ scal,
                           int n)
{
    const int i = blockIdx.x * 256 + threadIdx.x;
    if (i >= n) return;
    const float rtr = scal[0], den = scal[1];
    const float alpha = (rtr > CG_TOL) ? rtr / den : 0.f;
    X[i].x += alpha * Pv[i].x;  X[i].y += alpha * Pv[i].y;
    R[i].x -= alpha * Ap[i].x;  R[i].y -= alpha * Ap[i].y;
}

__global__ void cg_update2(float2* Pv, const float2* __restrict__ R,
                           const float* __restrict__ scal, int n)
{
    const int i = blockIdx.x * 256 + threadIdx.x;
    if (i >= n) return;
    const float rtr = scal[0];
    if (rtr > CG_TOL) {
        const float beta = scal[2] / rtr;
        Pv[i] = make_float2(R[i].x + beta * Pv[i].x, R[i].y + beta * Pv[i].y);
    }
}

__global__ void advance_scal(float* s)
{
    if (threadIdx.x == 0 && s[0] > CG_TOL) s[0] = s[2];
}

__global__ void write_out(const float2* __restrict__ X, float* __restrict__ out, int n)
{
    const int i = blockIdx.x * 256 + threadIdx.x;
    if (i >= n) return;
    const int b = i >> 16, yx = i & 65535;
    out[(((size_t)b * 2 + 0) << 16) + yx] = X[i].x;
    out[(((size_t)b * 2 + 1) << 16) + yx] = X[i].y;
}

// ---------------------------------------------------------------------------
extern "C" void kernel_launch(void* const* d_in, const int* in_sizes, int n_in,
                              void* d_out, int out_size, void* d_ws, size_t ws_size,
                              hipStream_t stream)
{
    (void)in_sizes; (void)n_in; (void)out_size; (void)ws_size;
    const float* atb  = (const float*)d_in[0];
    const float* csr  = (const float*)d_in[1];
    const float* csi  = (const float*)d_in[2];
    const float* mask = (const float*)d_in[3];
    const float* w1 = (const float*)d_in[4];  const float* b1 = (const float*)d_in[5];
    const float* w2 = (const float*)d_in[6];  const float* b2 = (const float*)d_in[7];
    const float* w3 = (const float*)d_in[8];  const float* b3 = (const float*)d_in[9];
    const float* w4 = (const float*)d_in[10]; const float* b4 = (const float*)d_in[11];
    const float* w5 = (const float*)d_in[12]; const float* b5 = (const float*)d_in[13];
    const float* lam = (const float*)d_in[14];

    char* ws = (char*)d_ws;
    float*          scal = (float*)ws;                                   // [16]
    unsigned short* wrep = (unsigned short*)(ws + 65536);                // 3*36864 bf16
    unsigned short* A0   = (unsigned short*)(ws + (1u << 20));           // 64 MB bf16 NHWC
    unsigned short* A1   = A0 + (size_t)NPIX * 64;                       // 64 MB bf16 NHWC
    float2* COIL = (float2*)(ws + (1u << 20));                           // 48 MB (reuse A0)
    float2* RHS  = (float2*)(ws + (1u << 20) + (48u << 20));             // 4 MB
    float2* X    = (float2*)(ws + (1u << 20) + (64u << 20));             // reuse A1 region
    float2* R    = X + NPIX;
    float2* Pv   = R + NPIX;
    float2* Ap   = Pv + NPIX;

    // ---- CNN denoiser ----
    repack_w<<<432, 256, 0, stream>>>(w2, w3, w4, wrep);
    conv1_in<<<NPIX / 256, 256, 0, stream>>>(atb, w1, b1, A0);
    const size_t clds = (size_t)3 * 130 * 64 * sizeof(unsigned short);   // 49920 B
    conv64_wmma<<<dim3(2, HH, BB), 256, clds, stream>>>(A0, wrep,             b2, A1, 1);
    conv64_wmma<<<dim3(2, HH, BB), 256, clds, stream>>>(A1, wrep + 36864,     b3, A0, 1);
    conv64_wmma<<<dim3(2, HH, BB), 256, clds, stream>>>(A0, wrep + 2 * 36864, b4, A1, 1);
    conv5_rhs<<<NPIX / 256, 256, 0, stream>>>(A1, atb, w5, b5, lam, RHS);

    // ---- CG init ----
    cg_init<<<NPIX / 256, 256, 0, stream>>>(RHS, X, R, Pv, NPIX);
    zero_scal<<<1, 32, 0, stream>>>(scal, 0);
    reduce_norm<<<512, 256, 0, stream>>>(R, NPIX, scal + 0);

    // ---- 11 fixed CG iterations ----
    for (int it = 0; it < 11; ++it) {
        coil_rowfft<<<dim3(32, CC, BB), 256, 0, stream>>>(Pv, csr, csi, COIL);
        colfft_mask<<<dim3(32, CC, BB), 256, 0, stream>>>(COIL, mask);
        rowifft_accum<<<dim3(32, BB), 256, 0, stream>>>(COIL, csr, csi, Pv, lam, Ap);
        zero_scal<<<1, 32, 0, stream>>>(scal, 1);
        reduce_dot<<<512, 256, 0, stream>>>(Pv, Ap, NPIX, scal + 1);
        cg_update1<<<NPIX / 256, 256, 0, stream>>>(X, R, Pv, Ap, scal, NPIX);
        zero_scal<<<1, 32, 0, stream>>>(scal, 2);
        reduce_norm<<<512, 256, 0, stream>>>(R, NPIX, scal + 2);
        cg_update2<<<NPIX / 256, 256, 0, stream>>>(Pv, R, scal, NPIX);
        advance_scal<<<1, 32, 0, stream>>>(scal);
    }

    write_out<<<NPIX / 256, 256, 0, stream>>>(X, (float*)d_out, NPIX);
}
